// DenseRnn_48387101556808
// MI455X (gfx1250) — matrix operand order repaired and checked by reading, not compiled, not run on hardware
//
#include <hip/hip_runtime.h>
#include <hip/hip_bf16.h>

// ---------------------------------------------------------------------------
// DenseRNN (gated DeltaNet-style) for MI455X / gfx1250, wave32 + WMMA.
// B=2, N=2048, D=1024, H=8, HD=128.
// ---------------------------------------------------------------------------

typedef __attribute__((ext_vector_type(16))) _Float16 v16h;
typedef __attribute__((ext_vector_type(8)))  _Float16 v8h;
typedef __attribute__((ext_vector_type(8)))  float    v8f;
typedef __attribute__((ext_vector_type(4)))  float    f4v;

#define EPSF 1e-6f

// ------------------------------ converts -----------------------------------

__global__ void f32_to_f16_kernel(const float* __restrict__ in,
                                  _Float16* __restrict__ out, int n) {
  int i = blockIdx.x * blockDim.x + threadIdx.x;
  if (i < n) out[i] = (_Float16)in[i];
}

// in: fp32 [R][C] row-major  ->  out: f16 [C][R]  (i.e. W^T with rows=N, cols=K)
__global__ void transpose_f32_to_f16_kernel(const float* __restrict__ in,
                                            _Float16* __restrict__ out,
                                            int R, int C) {
  int i = blockIdx.x * blockDim.x + threadIdx.x;
  if (i >= R * C) return;
  int r = i / C, c = i % C;
  out[(size_t)c * R + r] = (_Float16)in[i];
}

// ------------------------------ WMMA GEMM ----------------------------------
// C[M][Nn] (f32) = act( A[M][K] (f16, row-major) @ Bt[Nn][K]^T )
// One 16x64 C strip per wave (1 A frag, 4 B frags, 4 WMMAs per k-step),
// 8 waves per block. All 4 B fragments are loaded into distinct registers
// BEFORE the WMMAs so s_wait_loadcnt can stagger instead of draining to 0.
// act: 0=none, 1=silu, 2=log-sigmoid, 3=sigmoid.
__global__ __launch_bounds__(256) void wmma_gemm_kernel(
    const _Float16* __restrict__ A, const _Float16* __restrict__ Bt,
    float* __restrict__ C, int M, int Nn, int K, int act) {
  int wave = threadIdx.x >> 5;
  int lane = threadIdx.x & 31;
  int tilesN = Nn >> 6;                     // 64-wide strips
  int tile = blockIdx.x * 8 + wave;
  int tm = tile / tilesN;
  int tn = tile % tilesN;

  int rA  = tm * 16 + (lane & 15);          // A row for this lane
  int kbA = (lane & 16) ? 16 : 0;           // contiguous k = 16h + s, matches B
  int kbB = (lane & 16) ? 16 : 0;           // contiguous k = 16h + s, matches A

  const _Float16* Ar = A + (size_t)rA * K;
  const _Float16* Br0 = Bt + (size_t)(tn * 64 + (lane & 15)) * K + kbB;

  v8f acc[4] = {};
#pragma unroll 2
  for (int k0 = 0; k0 < K; k0 += 32) {
    v8h alo = *(const v8h*)(Ar + k0 + kbA);
    v8h ahi = *(const v8h*)(Ar + k0 + kbA + 8);
    v16h av;
#pragma unroll
    for (int h = 0; h < 8; ++h) { av[h] = alo[h]; av[h + 8] = ahi[h]; }
    v16h bvs[4];
#pragma unroll
    for (int s = 0; s < 4; ++s) {
      bvs[s] = *(const v16h*)(Br0 + (size_t)(s * 16) * K + k0);
    }
#pragma unroll
    for (int s = 0; s < 4; ++s) {
      acc[s] = __builtin_amdgcn_wmma_f32_16x16x32_f16(
          /*neg_a=*/false, av, /*neg_b=*/false, bvs[s],
          /*c_mod=*/(short)0, acc[s], /*reuse_a=*/false, /*reuse_b=*/false);
    }
  }

  int rowAdd = (lane & 16) ? 8 : 0;         // C layout: VGPR v -> row v(+8)
  float* Cp = C + (size_t)(tm * 16 + rowAdd) * Nn + tn * 64 + (lane & 15);
#pragma unroll
  for (int s = 0; s < 4; ++s) {
#pragma unroll
    for (int v = 0; v < 8; ++v) {
      float x = acc[s][v];
      if (act == 1) {
        x = x / (1.f + expf(-x));                               // silu
      } else if (act == 2) {
        x = fminf(x, 0.f) - log1pf(expf(-fabsf(x)));            // log-sigmoid
      } else if (act == 3) {
        x = 1.f / (1.f + expf(-x));                             // sigmoid
      }
      Cp[(size_t)v * Nn + s * 16] = x;
    }
  }
}

// ------------------------------ beta ---------------------------------------
// Beta[row][h] = 2*sigmoid( x_row . Wbeta[:,h] )
__global__ void beta_kernel(const float* __restrict__ X,
                            const float* __restrict__ Wb,
                            float* __restrict__ Beta,
                            int rows, int D, int H) {
  int id = blockIdx.x * blockDim.x + threadIdx.x;
  if (id >= rows * H) return;
  int row = id / H, h = id % H;
  const float* xr = X + (size_t)row * D;
  float acc = 0.f;
  for (int i = 0; i < D; ++i) acc += xr[i] * Wb[i * H + h];
  Beta[id] = 2.f / (1.f + expf(-acc));
}

// ------------------------------ scan ---------------------------------------
// One (b,h) per block; 256 threads. Lane tid owns HALF of column j = tid&127:
// rows [half*64, half*64+64) of the 128x128 state S, kept in 64 registers
// per lane (small enough that full unroll + SROA is trivially profitable --
// no scratch spill). Column sums (c0/c1/o) are 2-way partials combined via a
// small LDS ping-pong buffer. ||k||^2 uses a wave32 shfl_xor reduction.
// Per-token broadcasts packed in LDS as float4 {k, e, e*beta*k, q}.
// Fused DPLR sub-steps (e = sigmoid(f) precomputed by GEMM epilogue):
//   c0 = k^T S
//   S1 = Diag(e) S - (e*beta*k) c0^T          ; c1 = k^T S1
//   S2 = S1 + k (v - beta*c1)^T               ; o  = q^T S2
__global__ __launch_bounds__(256, 1) void scan_kernel(
    const float* __restrict__ Q, const float* __restrict__ Kv,
    const float* __restrict__ V, const float* __restrict__ E,
    const float* __restrict__ Beta, float* __restrict__ O, int Ntok) {
  __shared__ f4v   pk[128];      // {k, e, e*beta*k, q} per row index i
  __shared__ float wred[4];      // per-wave partial of ||k||^2 (half 0 waves)
  __shared__ float part0[256];   // ping
  __shared__ float part1[256];   // pong
  __shared__ float pbeta;

  const int D = 1024;
  int bh = blockIdx.x;
  int b = bh >> 3;               // H = 8
  int h = bh & 7;
  int tid  = threadIdx.x;
  int j    = tid & 127;          // column
  int half = tid >> 7;           // 0: rows 0..63, 1: rows 64..127
  int rbase = half << 6;

  float S[64];
#pragma unroll
  for (int i = 0; i < 64; ++i) S[i] = 0.f;

  for (int t = 0; t < Ntok; ++t) {
    size_t row  = (size_t)b * Ntok + t;
    size_t base = row * D + h * 128;

    float vj = V[base + j];                  // needed by both halves
    float kr = 0.f, ej = 0.f, qj = 0.f;
    if (half == 0) {
      kr = Kv[base + j];
      ej = E[base + j];
      qj = Q[base + j];
    } else if (t + 1 < Ntok) {               // upper half prefetches next token
      __builtin_prefetch(Kv + base + D + j, 0, 0);
      __builtin_prefetch(E  + base + D + j, 0, 0);
      __builtin_prefetch(V  + base + D + j, 0, 0);
      __builtin_prefetch(Q  + base + D + j, 0, 0);
    }

    // ||k||^2: wave32 shuffle reduce (half-0 waves), then 4 partials in LDS
    float ss = kr * kr;
#pragma unroll
    for (int off = 16; off > 0; off >>= 1) ss += __shfl_xor(ss, off, 32);
    if (half == 0 && (tid & 31) == 0) wred[tid >> 5] = ss;
    if (tid == 0) pbeta = Beta[row * 8 + h];
    __syncthreads();

    float beta = pbeta;
    if (half == 0) {
      float tot = wred[0] + wred[1] + wred[2] + wred[3];
      float rn  = rsqrtf(tot + EPSF);        // l2norm(k)
      float kj  = kr * rn;
      f4v p4; p4[0] = kj; p4[1] = ej; p4[2] = ej * beta * kj; p4[3] = qj;
      pk[j] = p4;
    }
    __syncthreads();

    // ---- pass 0: c0 = k^T S ----
    float c0p = 0.f;
#pragma unroll
    for (int i = 0; i < 64; ++i) c0p += pk[rbase + i][0] * S[i];
    part0[tid] = c0p;
    __syncthreads();
    float nc0 = -(part0[j] + part0[j + 128]);

    // ---- pass 1: S1 = e*S + (e*beta*k)*nc0 ; c1 = k^T S1 ----
    float c1p = 0.f;
#pragma unroll
    for (int i = 0; i < 64; ++i) {
      f4v p = pk[rbase + i];
      float s1 = p[1] * S[i] + p[2] * nc0;
      S[i] = s1;
      c1p += p[0] * s1;
    }
    part1[tid] = c1p;
    __syncthreads();
    float c1 = part1[j] + part1[j + 128];

    // ---- pass 2: S2 = S1 + k*(v - beta*c1) ; o = q^T S2 ----
    float w  = vj - beta * c1;
    float ojp = 0.f;
#pragma unroll
    for (int i = 0; i < 64; ++i) {
      f4v p = pk[rbase + i];
      float s2 = S[i] + p[0] * w;
      S[i] = s2;
      ojp += p[3] * s2;
    }
    part0[tid] = ojp;                        // ping-pong: part0 safe to reuse
    __syncthreads();
    if (half == 0) O[base + j] = part0[j] + part0[j + 128];
    // next token's first barrier orders pk/part rewrites after all reads
  }
}

// ------------------------- gated RMSNorm (per head) -------------------------
// y = (o * sigmoid(gate)) / sqrt(mean_head(y^2)+eps) * norm_w  -> f16 for Wo GEMM
__global__ __launch_bounds__(128) void norm_gate_kernel(
    const float* __restrict__ O, const float* __restrict__ Gate,
    const float* __restrict__ normw, _Float16* __restrict__ Yh, int D) {
  __shared__ float red[128];
  int h = blockIdx.x;
  int t = blockIdx.y;
  int j = threadIdx.x;
  size_t idx = (size_t)t * D + h * 128 + j;
  float yg = O[idx] * (1.f / (1.f + expf(-Gate[idx])));
  red[j] = yg * yg;
  __syncthreads();
  for (int s = 64; s > 0; s >>= 1) {
    if (j < s) red[j] += red[j + s];
    __syncthreads();
  }
  float ms = red[0] * (1.f / 128.f);
  float y = yg * rsqrtf(ms + EPSF) * normw[h * 128 + j];
  Yh[idx] = (_Float16)y;
}

// ------------------------------ launcher -----------------------------------

extern "C" void kernel_launch(void* const* d_in, const int* in_sizes, int n_in,
                              void* d_out, int out_size, void* d_ws, size_t ws_size,
                              hipStream_t stream) {
  (void)in_sizes; (void)n_in; (void)out_size; (void)ws_size;
  const int Bsz = 2, Ntok = 2048, D = 1024, H = 8, HD = 128;
  const int M = Bsz * Ntok;               // 4096 rows

  const float* x     = (const float*)d_in[0];
  const float* Wq    = (const float*)d_in[1];
  const float* Wk    = (const float*)d_in[2];
  const float* Wv    = (const float*)d_in[3];
  const float* Wf1   = (const float*)d_in[4];
  const float* Wf2   = (const float*)d_in[5];
  const float* Wbeta = (const float*)d_in[6];
  const float* Wog1  = (const float*)d_in[7];
  const float* Wog2  = (const float*)d_in[8];
  const float* normw = (const float*)d_in[9];
  const float* Wo    = (const float*)d_in[10];
  float* Out = (float*)d_out;

  // ---- workspace carve-out (256B aligned) ----
  char* ws = (char*)d_ws;
  size_t off = 0;
  auto alloc = [&](size_t bytes) -> void* {
    void* p = ws + off;
    off = (off + bytes + 255) & ~(size_t)255;
    return p;
  };
  _Float16* Xh    = (_Float16*)alloc((size_t)M * D * 2);
  _Float16* Wqt   = (_Float16*)alloc((size_t)D * D * 2);
  _Float16* Wkt   = (_Float16*)alloc((size_t)D * D * 2);
  _Float16* Wvt   = (_Float16*)alloc((size_t)D * D * 2);
  _Float16* Wot   = (_Float16*)alloc((size_t)D * D * 2);
  _Float16* Wf1t  = (_Float16*)alloc((size_t)HD * D * 2);
  _Float16* Wf2t  = (_Float16*)alloc((size_t)D * HD * 2);
  _Float16* Wog1t = (_Float16*)alloc((size_t)HD * D * 2);
  _Float16* Wog2t = (_Float16*)alloc((size_t)D * HD * 2);
  float* Qf    = (float*)alloc((size_t)M * D * 4);
  float* Kf    = (float*)alloc((size_t)M * D * 4);
  float* Vf    = (float*)alloc((size_t)M * D * 4);
  float* Ef    = (float*)alloc((size_t)M * D * 4);   // sigmoid(f) = exp(log_f)
  float* Gatef = (float*)alloc((size_t)M * D * 4);
  float* Of    = (float*)alloc((size_t)M * D * 4);
  float* F1f   = (float*)alloc((size_t)M * HD * 4);
  float* G1f   = (float*)alloc((size_t)M * HD * 4);
  _Float16* F1h = (_Float16*)alloc((size_t)M * HD * 2);
  _Float16* G1h = (_Float16*)alloc((size_t)M * HD * 2);
  float* Betaf = (float*)alloc((size_t)M * H * 4);
  _Float16* Yh = (_Float16*)alloc((size_t)M * D * 2);

  // ---- 1. f16 copies / weight transposes ----
  int nx = M * D;
  f32_to_f16_kernel<<<nx / 256, 256, 0, stream>>>(x, Xh, nx);
  transpose_f32_to_f16_kernel<<<(D * D + 255) / 256, 256, 0, stream>>>(Wq,  Wqt,  D, D);
  transpose_f32_to_f16_kernel<<<(D * D + 255) / 256, 256, 0, stream>>>(Wk,  Wkt,  D, D);
  transpose_f32_to_f16_kernel<<<(D * D + 255) / 256, 256, 0, stream>>>(Wv,  Wvt,  D, D);
  transpose_f32_to_f16_kernel<<<(D * D + 255) / 256, 256, 0, stream>>>(Wo,  Wot,  D, D);
  transpose_f32_to_f16_kernel<<<(D * HD + 255) / 256, 256, 0, stream>>>(Wf1,  Wf1t,  D, HD);
  transpose_f32_to_f16_kernel<<<(HD * D + 255) / 256, 256, 0, stream>>>(Wf2,  Wf2t,  HD, D);
  transpose_f32_to_f16_kernel<<<(D * HD + 255) / 256, 256, 0, stream>>>(Wog1, Wog1t, D, HD);
  transpose_f32_to_f16_kernel<<<(HD * D + 255) / 256, 256, 0, stream>>>(Wog2, Wog2t, HD, D);

  // ---- 2. projections (WMMA, 16x64 strip per wave) ----
  int blkBig   = (M / 16) * (D / 64) / 8;   // 512
  int blkSmall = (M / 16) * (HD / 64) / 8;  // 64
  wmma_gemm_kernel<<<blkBig, 256, 0, stream>>>(Xh, Wqt, Qf, M, D, D, 1);   // silu
  wmma_gemm_kernel<<<blkBig, 256, 0, stream>>>(Xh, Wkt, Kf, M, D, D, 1);   // silu
  wmma_gemm_kernel<<<blkBig, 256, 0, stream>>>(Xh, Wvt, Vf, M, D, D, 1);   // silu

  wmma_gemm_kernel<<<blkSmall, 256, 0, stream>>>(Xh, Wf1t, F1f, M, HD, D, 0);
  f32_to_f16_kernel<<<(M * HD) / 256, 256, 0, stream>>>(F1f, F1h, M * HD);
  wmma_gemm_kernel<<<blkBig, 256, 0, stream>>>(F1h, Wf2t, Ef, M, D, HD, 3); // sigmoid

  wmma_gemm_kernel<<<blkSmall, 256, 0, stream>>>(Xh, Wog1t, G1f, M, HD, D, 0);
  f32_to_f16_kernel<<<(M * HD) / 256, 256, 0, stream>>>(G1f, G1h, M * HD);
  wmma_gemm_kernel<<<blkBig, 256, 0, stream>>>(G1h, Wog2t, Gatef, M, D, HD, 0);

  beta_kernel<<<(M * H + 255) / 256, 256, 0, stream>>>(x, Wbeta, Betaf, M, D, H);

  // ---- 3. sequential recurrent scan: 16 blocks = B*H, S in registers ----
  scan_kernel<<<Bsz * H, 256, 0, stream>>>(Qf, Kf, Vf, Ef, Betaf, Of, Ntok);

  // ---- 4. gated per-head RMSNorm -> f16 ----
  norm_gate_kernel<<<dim3(H, M), 128, 0, stream>>>(Of, Gatef, normw, Yh, D);

  // ---- 5. output projection (WMMA) ----
  wmma_gemm_kernel<<<blkBig, 256, 0, stream>>>(Yh, Wot, Out, M, D, D, 0);
}
